// GPT2Attention_79963701117168
// MI455X (gfx1250) — matrix operand order repaired and checked by reading, not compiled, not run on hardware
//
#include <hip/hip_runtime.h>
#include <hip/hip_bf16.h>

typedef __bf16 bf16_t;
typedef __attribute__((ext_vector_type(16))) __bf16 v16bf;
typedef __attribute__((ext_vector_type(8)))  __bf16 bf16x8;
typedef __attribute__((ext_vector_type(8)))  float  v8f;

// TDM descriptor argument vectors (clang-23 6-arg tensor_load_to_lds)
typedef __attribute__((ext_vector_type(4))) unsigned int uint32x4;
typedef __attribute__((ext_vector_type(8))) int          int32x8;
typedef __attribute__((ext_vector_type(4))) int          int32x4;

#if __has_builtin(__builtin_amdgcn_tensor_load_to_lds)
#define USE_TDM 1
#else
#define USE_TDM 0
#endif

#define DEV __device__ __forceinline__

static constexpr int B_ = 4, S_ = 2304, E_ = 1024, H_ = 16, D_ = 64;
static constexpr int QKV_N = 3 * E_;          // 3072
static constexpr int MROWS = B_ * S_;          // 9216
static constexpr float NEGM  = -10000.0f;
static constexpr float SCALE = 0.125f;         // 1/sqrt(64)

// ---------------- WMMA fragment helpers (CDNA5 §7.12.2 layouts) ----------------

// A-matrix 16x32 bf16, source row-major with leading dim `ld` (elements).
DEV v16bf load_a_frag(const bf16_t* base, int ld) {
  const int lane = threadIdx.x & 31;
  const bf16_t* p = base + (size_t)(lane & 15) * ld + ((lane < 16) ? 0 : 8);
  bf16x8 lo = *(const bf16x8*)p;
  bf16x8 hi = *(const bf16x8*)(p + 16);
  v16bf a;
#pragma unroll
  for (int i = 0; i < 8; ++i) { a[i] = lo[i]; a[i + 8] = hi[i]; }
  return a;
}

// B-matrix 32x16 bf16 from N-major storage (row n holds K-contiguous values).
DEV v16bf load_b_frag(const bf16_t* base, int ld) {
  const int lane = threadIdx.x & 31;
  const bf16_t* p = base + (size_t)(lane & 15) * ld + ((lane < 16) ? 0 : 8);
  bf16x8 lo = *(const bf16x8*)p;
  bf16x8 hi = *(const bf16x8*)(p + 16);
  v16bf b;
#pragma unroll
  for (int i = 0; i < 8; ++i) { b[i] = lo[i]; b[i + 8] = hi[i]; }
  return b;
}

DEV v8f wmma_bf16(v16bf a, v16bf b, v8f c) {
  return __builtin_amdgcn_wmma_f32_16x16x32_bf16(false, a, false, b, (short)0, c,
                                                 false, false);
}

DEV v8f zero8() {
  v8f z;
#pragma unroll
  for (int i = 0; i < 8; ++i) z[i] = 0.0f;
  return z;
}

// ---------------- precision conversion kernels ----------------

__global__ __launch_bounds__(256) void k_f32_to_bf16(const float* __restrict__ x,
                                                     bf16_t* __restrict__ y, long n) {
  long i = ((long)blockIdx.x * 256 + threadIdx.x) * 8;
  if (i + 8 > n) return;
  float4 a = *(const float4*)(x + i);
  float4 b = *(const float4*)(x + i + 4);
  bf16x8 o;
  o[0] = (bf16_t)a.x; o[1] = (bf16_t)a.y; o[2] = (bf16_t)a.z; o[3] = (bf16_t)a.w;
  o[4] = (bf16_t)b.x; o[5] = (bf16_t)b.y; o[6] = (bf16_t)b.z; o[7] = (bf16_t)b.w;
  *(bf16x8*)(y + i) = o;
}

// W: K x N (f32) -> Wt: N x K (bf16), LDS-tiled transpose
__global__ __launch_bounds__(256) void k_transpose_bf16(const float* __restrict__ W,
                                                        bf16_t* __restrict__ Wt,
                                                        int K, int N) {
  __shared__ float tile[32][33];
  const int tx = threadIdx.x & 31, ty = threadIdx.x >> 5;
  const int k0 = blockIdx.y * 32, n0 = blockIdx.x * 32;
#pragma unroll
  for (int i = 0; i < 4; ++i)
    tile[ty + i * 8][tx] = W[(size_t)(k0 + ty + i * 8) * N + n0 + tx];
  __syncthreads();
#pragma unroll
  for (int i = 0; i < 4; ++i)
    Wt[(size_t)(n0 + ty + i * 8) * K + k0 + tx] = (bf16_t)tile[tx][ty + i * 8];
}

// ---------------- QKV GEMM: (9216x1024)x(1024x3072)+bias, scatter heads ----------------

__global__ __launch_bounds__(256) void k_gemm_qkv(const bf16_t* __restrict__ A,
                                                  const bf16_t* __restrict__ Wt,
                                                  const float* __restrict__ bias,
                                                  bf16_t* __restrict__ qb,
                                                  bf16_t* __restrict__ kb,
                                                  bf16_t* __restrict__ vb) {
  const int K = E_;
  const int lane = threadIdx.x & 31, wid = threadIdx.x >> 5;
  const int wy = wid & 1, wx = wid >> 1;
  const int mbase = blockIdx.y * 64 + wy * 32;
  const int nbase = blockIdx.x * 128 + wx * 32;
  v8f c00 = zero8(), c01 = zero8(), c10 = zero8(), c11 = zero8();

  for (int kk = 0; kk < K; kk += 32) {
    __builtin_prefetch(A + (size_t)mbase * K + kk + 64, 0, 1);
    __builtin_prefetch(Wt + (size_t)nbase * K + kk + 64, 0, 1);
    v16bf a0 = load_a_frag(A + (size_t)mbase * K + kk, K);
    v16bf a1 = load_a_frag(A + (size_t)(mbase + 16) * K + kk, K);
    v16bf b0 = load_b_frag(Wt + (size_t)nbase * K + kk, K);
    v16bf b1 = load_b_frag(Wt + (size_t)(nbase + 16) * K + kk, K);
    c00 = wmma_bf16(a0, b0, c00);
    c01 = wmma_bf16(a0, b1, c01);
    c10 = wmma_bf16(a1, b0, c10);
    c11 = wmma_bf16(a1, b1, c11);
  }

  const int hi16 = lane >> 4;
#pragma unroll
  for (int mi = 0; mi < 2; ++mi)
#pragma unroll
    for (int ni = 0; ni < 2; ++ni) {
      const v8f& c = mi ? (ni ? c11 : c10) : (ni ? c01 : c00);
#pragma unroll
      for (int r = 0; r < 8; ++r) {
        int m = mbase + mi * 16 + r + 8 * hi16;
        int n = nbase + ni * 16 + (lane & 15);
        float v = c[r] + bias[n];
        int part = n >> 10, e = n & 1023, h = e >> 6, dd = e & 63;
        int b = m / S_, s = m - b * S_;
        bf16_t* dst = (part == 0) ? qb : (part == 1) ? kb : vb;
        dst[(((size_t)(b * H_ + h)) * S_ + s) * D_ + dd] = (bf16_t)v;
      }
    }
}

// ---------------- sparse flash attention ----------------
// grid: (S/128, H, B); block 256 = 8 waves; each wave owns 16 query rows.
// keys = 72 strided (rows 31,63,...,2303) ++ local block (512 or 256 rows), causal.
// K tile staged by the Tensor Data Mover in gather mode (16 row indices / D#),
// V tile staged manually (needs transpose for the PV B-fragment).

__global__ __launch_bounds__(256) void k_attn(const bf16_t* __restrict__ qb,
                                              const bf16_t* __restrict__ kb,
                                              const bf16_t* __restrict__ vb,
                                              bf16_t* __restrict__ ob) {
  __shared__ bf16_t ldsK[32][64];        // [key][d]      4 KB  (TDM gather target)
  __shared__ bf16_t ldsVt[64][32];       // [d][key]      4 KB
  __shared__ bf16_t ldsP[8][16][32];     // per-wave P    8 KB

  const int lane = threadIdx.x & 31, wid = threadIdx.x >> 5;
  const int hi16 = lane >> 4;
  const int b = blockIdx.z, h = blockIdx.y;
  const int q0 = blockIdx.x * 128;
  const int qrow0 = q0 + wid * 16;
  const int local0 = (q0 < 2048) ? (q0 & ~511) : 2048;
  const int nloc = (q0 < 2048) ? 512 : 256;
  const int nkeys = 72 + nloc;
  const int npairs = (nkeys + 31) / 32;
  const size_t bh = ((size_t)b * H_ + h) * S_;

  v16bf qf0 = load_a_frag(qb + (bh + qrow0) * D_ + 0, D_);
  v16bf qf1 = load_a_frag(qb + (bh + qrow0) * D_ + 32, D_);

  v8f o0 = zero8(), o1 = zero8(), o2 = zero8(), o3 = zero8();
  float rmax[8], rsum[8];
#pragma unroll
  for (int r = 0; r < 8; ++r) { rmax[r] = -3.0e38f; rsum[r] = 0.0f; }

  const int skey = threadIdx.x >> 3;        // 0..31
  const int sd   = (threadIdx.x & 7) * 8;   // 0..56

  for (int t = 0; t < npairs; ++t) {
    __syncthreads();  // previous tile fully consumed

#if USE_TDM
    if (wid == 0) {
      // Two 16-row TDM gathers stage the 32-key K tile: D# per CDNA5 §8.3-8.7.
      const unsigned long long gaddr =
          (unsigned long long)(size_t)(kb + bh * (size_t)D_);
#pragma unroll
      for (int halfg = 0; halfg < 2; ++halfg) {
        unsigned idx[16];
#pragma unroll
        for (int i = 0; i < 16; ++i) {
          int j = t * 32 + halfg * 16 + i;
          int g = (j < nkeys) ? ((j < 72) ? j * 32 + 31 : local0 + (j - 72)) : 0;
          idx[i] = (unsigned)g;
        }
        uint32x4 g0;
        g0[0] = 0x80000001u;  // count=1, is_restore=0, 16b indices, gather_mode=1
        g0[1] = (unsigned)(size_t)&ldsK[halfg * 16][0];       // lds_addr
        g0[2] = (unsigned)gaddr;                              // global_addr[31:0]
        g0[3] = (unsigned)((gaddr >> 32) & 0x01FFFFFFu) | (2u << 30);  // type=2
        int32x8 g1;
        g1[0] = (int)(1u << 16);                 // data_size = 2 bytes
        g1[1] = (int)(64u << 16);                // tensor_dim0 = 64 (D)
        g1[2] = (int)((unsigned)S_ << 16);       // tensor_dim1 = 2304 (S)
        g1[3] = (int)(64u << 16);                // tile_dim0 = 64
        g1[4] = 16;                              // tile_dim1 = #valid indices
        g1[5] = 64;                              // tensor_dim0_stride = 64
        g1[6] = 0;
        g1[7] = 0;
        int32x4 g2, g3;
#pragma unroll
        for (int i = 0; i < 4; ++i)
          g2[i] = (int)(idx[2 * i] | (idx[2 * i + 1] << 16));
#pragma unroll
        for (int i = 0; i < 4; ++i)
          g3[i] = (int)(idx[8 + 2 * i] | (idx[8 + 2 * i + 1] << 16));
        int32x8 gpad;  // trailing descriptor block (unused): zero-filled
#pragma unroll
        for (int i = 0; i < 8; ++i) gpad[i] = 0;
        __builtin_amdgcn_tensor_load_to_lds(g0, g1, g2, g3, gpad, 0);
      }
    }
#endif

    {
      int j = t * 32 + skey;
      bf16x8 vv;
#if !USE_TDM
      bf16x8 kk;
#endif
      if (j < nkeys) {
        int g = (j < 72) ? (j * 32 + 31) : (local0 + (j - 72));
        vv = *(const bf16x8*)(vb + (bh + g) * D_ + sd);
#if !USE_TDM
        kk = *(const bf16x8*)(kb + (bh + g) * D_ + sd);
#endif
      } else {
#pragma unroll
        for (int i = 0; i < 8; ++i) vv[i] = (bf16_t)0.0f;
#if !USE_TDM
#pragma unroll
        for (int i = 0; i < 8; ++i) kk[i] = (bf16_t)0.0f;
#endif
      }
#if !USE_TDM
      *(bf16x8*)&ldsK[skey][sd] = kk;
#endif
#pragma unroll
      for (int i = 0; i < 8; ++i) ldsVt[sd + i][skey] = vv[i];
    }

#if USE_TDM
    if (wid == 0) {
#if __has_builtin(__builtin_amdgcn_s_wait_tensorcnt)
      __builtin_amdgcn_s_wait_tensorcnt(0);
#endif
    }
#endif
    __syncthreads();  // tile staged (TDM K + manual V^T)

    // scores: Q(16x64) x K^T for two 16-key halves
    v8f s0 = zero8(), s1 = zero8();
    s0 = wmma_bf16(qf0, load_b_frag(&ldsK[0][0], 64), s0);
    s0 = wmma_bf16(qf1, load_b_frag(&ldsK[0][32], 64), s0);
    s1 = wmma_bf16(qf0, load_b_frag(&ldsK[16][0], 64), s1);
    s1 = wmma_bf16(qf1, load_b_frag(&ldsK[16][32], 64), s1);

    // mask + online softmax (rows live in lanes 0-15 / 16-31 halves)
#pragma unroll
    for (int r = 0; r < 8; ++r) {
      int mrow = qrow0 + r + 8 * hi16;
      float sv0, sv1;
      {
        int j = t * 32 + (lane & 15);
        int g = (j < 72) ? (j * 32 + 31) : (local0 + (j - 72));
        bool ok = (j < nkeys) && (g <= mrow);
        sv0 = s0[r] * SCALE + (ok ? 0.0f : NEGM);
      }
      {
        int j = t * 32 + 16 + (lane & 15);
        int g = (j < 72) ? (j * 32 + 31) : (local0 + (j - 72));
        bool ok = (j < nkeys) && (g <= mrow);
        sv1 = s1[r] * SCALE + (ok ? 0.0f : NEGM);
      }
      float mx = fmaxf(sv0, sv1);
#pragma unroll
      for (int off = 8; off; off >>= 1) mx = fmaxf(mx, __shfl_xor(mx, off, 32));
      float nm = fmaxf(rmax[r], mx);
      float al = __expf(rmax[r] - nm);
      float p0 = __expf(sv0 - nm);
      float p1 = __expf(sv1 - nm);
      float ps = p0 + p1;
#pragma unroll
      for (int off = 8; off; off >>= 1) ps += __shfl_xor(ps, off, 32);
      rsum[r] = rsum[r] * al + ps;
      rmax[r] = nm;
      ldsP[wid][r + 8 * hi16][(lane & 15)]      = (bf16_t)p0;
      ldsP[wid][r + 8 * hi16][16 + (lane & 15)] = (bf16_t)p1;
      o0[r] *= al; o1[r] *= al; o2[r] *= al; o3[r] *= al;
    }

    // O += P(16x32) x V(32x64); P re-layouted through wave-private LDS
    v16bf pf = load_a_frag(&ldsP[wid][0][0], 32);
    o0 = wmma_bf16(pf, load_b_frag(&ldsVt[0][0], 32), o0);
    o1 = wmma_bf16(pf, load_b_frag(&ldsVt[16][0], 32), o1);
    o2 = wmma_bf16(pf, load_b_frag(&ldsVt[32][0], 32), o2);
    o3 = wmma_bf16(pf, load_b_frag(&ldsVt[48][0], 32), o3);
  }

  // normalize + write [B][S][E] bf16 (feeds proj GEMM as row-major A)
#pragma unroll
  for (int dt = 0; dt < 4; ++dt) {
    const v8f& o = dt == 0 ? o0 : dt == 1 ? o1 : dt == 2 ? o2 : o3;
#pragma unroll
    for (int r = 0; r < 8; ++r) {
      int mrow = qrow0 + r + 8 * hi16;
      float v = o[r] / rsum[r];
      ob[((size_t)(b * S_ + mrow)) * E_ + h * D_ + dt * 16 + (lane & 15)] = (bf16_t)v;
    }
  }
}

// ---------------- output projection: (9216x1024)x(1024x1024)+bias -> f32 ----------------

__global__ __launch_bounds__(256) void k_gemm_proj(const bf16_t* __restrict__ A,
                                                   const bf16_t* __restrict__ Wt,
                                                   const float* __restrict__ bias,
                                                   float* __restrict__ out) {
  const int K = E_, N = E_;
  const int lane = threadIdx.x & 31, wid = threadIdx.x >> 5;
  const int wy = wid & 1, wx = wid >> 1;
  const int mbase = blockIdx.y * 64 + wy * 32;
  const int nbase = blockIdx.x * 128 + wx * 32;
  v8f c00 = zero8(), c01 = zero8(), c10 = zero8(), c11 = zero8();

  for (int kk = 0; kk < K; kk += 32) {
    __builtin_prefetch(A + (size_t)mbase * K + kk + 64, 0, 1);
    __builtin_prefetch(Wt + (size_t)nbase * K + kk + 64, 0, 1);
    v16bf a0 = load_a_frag(A + (size_t)mbase * K + kk, K);
    v16bf a1 = load_a_frag(A + (size_t)(mbase + 16) * K + kk, K);
    v16bf b0 = load_b_frag(Wt + (size_t)nbase * K + kk, K);
    v16bf b1 = load_b_frag(Wt + (size_t)(nbase + 16) * K + kk, K);
    c00 = wmma_bf16(a0, b0, c00);
    c01 = wmma_bf16(a0, b1, c01);
    c10 = wmma_bf16(a1, b0, c10);
    c11 = wmma_bf16(a1, b1, c11);
  }

  const int hi16 = lane >> 4;
#pragma unroll
  for (int mi = 0; mi < 2; ++mi)
#pragma unroll
    for (int ni = 0; ni < 2; ++ni) {
      const v8f& c = mi ? (ni ? c11 : c10) : (ni ? c01 : c00);
#pragma unroll
      for (int r = 0; r < 8; ++r) {
        int m = mbase + mi * 16 + r + 8 * hi16;
        int n = nbase + ni * 16 + (lane & 15);
        out[(size_t)m * N + n] = c[r] + bias[n];
      }
    }
}

// ---------------- host launch ----------------

extern "C" void kernel_launch(void* const* d_in, const int* in_sizes, int n_in,
                              void* d_out, int out_size, void* d_ws, size_t ws_size,
                              hipStream_t stream) {
  (void)in_sizes; (void)n_in; (void)out_size; (void)ws_size;
  const float* hidden = (const float*)d_in[0];
  const float* wqkv   = (const float*)d_in[1];
  const float* bqkv   = (const float*)d_in[2];
  const float* wproj  = (const float*)d_in[3];
  const float* bproj  = (const float*)d_in[4];
  float* out = (float*)d_out;

  char* ws = (char*)d_ws;
  size_t off = 0;
  bf16_t* hid    = (bf16_t*)(ws + off); off += (size_t)MROWS * E_ * 2;
  bf16_t* wqkvT  = (bf16_t*)(ws + off); off += (size_t)QKV_N * E_ * 2;
  bf16_t* wprojT = (bf16_t*)(ws + off); off += (size_t)E_ * E_ * 2;
  bf16_t* qb     = (bf16_t*)(ws + off); off += (size_t)B_ * H_ * S_ * D_ * 2;
  bf16_t* kb     = (bf16_t*)(ws + off); off += (size_t)B_ * H_ * S_ * D_ * 2;
  bf16_t* vb     = (bf16_t*)(ws + off); off += (size_t)B_ * H_ * S_ * D_ * 2;
  bf16_t* attn   = (bf16_t*)(ws + off); off += (size_t)MROWS * E_ * 2;

  k_f32_to_bf16<<<(MROWS * E_) / 2048, 256, 0, stream>>>(hidden, hid, (long)MROWS * E_);
  k_transpose_bf16<<<dim3(QKV_N / 32, E_ / 32), 256, 0, stream>>>(wqkv, wqkvT, E_, QKV_N);
  k_transpose_bf16<<<dim3(E_ / 32, E_ / 32), 256, 0, stream>>>(wproj, wprojT, E_, E_);
  k_gemm_qkv<<<dim3(QKV_N / 128, MROWS / 64), 256, 0, stream>>>(hid, wqkvT, bqkv, qb, kb, vb);
  k_attn<<<dim3(S_ / 128, H_, B_), 256, 0, stream>>>(qb, kb, vb, attn);
  k_gemm_proj<<<dim3(E_ / 128, MROWS / 64), 256, 0, stream>>>(attn, wprojT, bproj, out);
}
